// EdgeFeatFilm_79517024518197
// MI455X (gfx1250) — matrix operand order repaired and checked by reading, not compiled, not run on hardware
//
#include <hip/hip_runtime.h>

typedef __attribute__((ext_vector_type(16))) __bf16 v16bf;
typedef __attribute__((ext_vector_type(8)))  float  v8f;

#define EDGE_B 16
#define EDGE_N 160
#define EDGE_D 1024
#define EDGE_E 128

// ---------------------------------------------------------------------------
// Pack a row-major fp32 weight matrix W[K x 128] into the wave32 WMMA
// B-matrix (32x16 bf16 per fragment) layout, split into bf16 hi + residual lo.
// Fragment index: ((ks*8 + ot)*32 + lane)*16 + e
//   K = ks*32 + (e>>3)*16 + (lane>>4)*8 + (e&7) ; n = ot*16 + (lane&15)
// ---------------------------------------------------------------------------
__global__ void k_pack_b(const float* __restrict__ W,
                         __bf16* __restrict__ hi, __bf16* __restrict__ lo) {
    int idx  = blockIdx.x * blockDim.x + threadIdx.x;   // exact coverage
    int e    = idx & 15;
    int lane = (idx >> 4) & 31;
    int ot   = (idx >> 9) & 7;
    int ks   = idx >> 12;
    int K    = ks * 32 + (e >> 3) * 16 + (lane >> 4) * 8 + (e & 7);
    int n    = ot * 16 + (lane & 15);
    float v  = W[(size_t)K * EDGE_E + n];
    __bf16 h = (__bf16)v;
    hi[idx]  = h;
    lo[idx]  = (__bf16)(v - (float)h);
}

// ---------------------------------------------------------------------------
// gamma/beta: gb = cond @ W_cond + b_cond ; store gamma+1 in [0:128], beta in
// [128:256] per batch. 16 blocks x 256 threads, one output each (K=1024 dot).
// ---------------------------------------------------------------------------
__global__ void k_cond(const float* __restrict__ cond, const float* __restrict__ Wc,
                       const float* __restrict__ bc, float* __restrict__ gb) {
    int b = blockIdx.x, c = threadIdx.x;
    const float* crow = cond + (size_t)b * EDGE_D;
    float s = 0.f;
    for (int k = 0; k < EDGE_D; ++k) s += crow[k] * Wc[(size_t)k * 256 + c];
    s += bc[c];
    if (c < EDGE_E) s += 1.0f;      // gamma = gb + 1
    gb[(size_t)b * 256 + c] = s;
}

// ---------------------------------------------------------------------------
// h = relu(node @ W_proj + b_proj) : M=2560, K=1024, N=128.
// bf16 WMMA with hi/lo split of A and B (3 WMMAs per K-step) for fp32-class
// accuracy. One 16x16 tile per wave; 160 blocks x 8 waves.
// ---------------------------------------------------------------------------
__global__ void __launch_bounds__(256, 1)
k_proj(const float* __restrict__ node, const float* __restrict__ bp,
       const __bf16* __restrict__ wph, const __bf16* __restrict__ wpl,
       float* __restrict__ h) {
    const int mtile = blockIdx.x;          // 0..159
    const int wave  = threadIdx.x >> 5;    // 0..7 -> n-tile
    const int lane  = threadIdx.x & 31;
    const int hf    = lane >> 4;
    const int lr    = lane & 15;
    const float* arow = node + (size_t)(mtile * 16 + lr) * EDGE_D;

    v8f acc = {};
    #pragma unroll 4
    for (int ks = 0; ks < EDGE_D / 32; ++ks) {
        // A fragment (16-bit A 16x32 layout): e<8 -> K=ks*32+hf*8+e, e>=8 -> +16
        int c0 = ks * 32 + hf * 8, c1 = c0 + 16;
        v16bf ah, al;
        #pragma unroll
        for (int e = 0; e < 8; ++e) {
            float v0 = arow[c0 + e], v1 = arow[c1 + e];
            __bf16 h0 = (__bf16)v0, h1 = (__bf16)v1;
            ah[e] = h0;     al[e]     = (__bf16)(v0 - (float)h0);
            ah[e + 8] = h1; al[e + 8] = (__bf16)(v1 - (float)h1);
        }
        size_t boff = ((size_t)(ks * 8 + wave) * 32 + lane) * 16;
        v16bf bh = *(const v16bf*)(wph + boff);
        v16bf bl = *(const v16bf*)(wpl + boff);
        acc = __builtin_amdgcn_wmma_f32_16x16x32_bf16(false, ah, false, bh, (short)0, acc, false, false);
        acc = __builtin_amdgcn_wmma_f32_16x16x32_bf16(false, ah, false, bl, (short)0, acc, false, false);
        acc = __builtin_amdgcn_wmma_f32_16x16x32_bf16(false, al, false, bh, (short)0, acc, false, false);
    }
    // C layout: element r of lane -> M = r + 8*hf, N = lr
    int n = wave * 16 + lr;
    float bias = bp[n];
    #pragma unroll
    for (int r = 0; r < 8; ++r) {
        int m = mtile * 16 + r + 8 * hf;
        float v = acc[r] + bias;
        h[(size_t)m * EDGE_E + n] = v > 0.f ? v : 0.f;
    }
}

// ---------------------------------------------------------------------------
// Main fused kernel: one block per (b,i); wave w owns j-tile w (16 rows x all
// 128 outputs). A[j,c] = h_i[c]*h_j[c] built in registers (hi/lo bf16 split),
// spatial feats folded in as one zero-padded WMMA K-step, then LayerNorm via
// shfl_xor (rows live in 16-lane halves), FiLM, ReLU, store.
// ---------------------------------------------------------------------------
__global__ void __launch_bounds__(320, 1)
k_edge(const float* __restrict__ h, const float* __restrict__ spat,
       const float* __restrict__ Wf, const float* __restrict__ bfm,
       const float* __restrict__ gb, const __bf16* __restrict__ wfh,
       const __bf16* __restrict__ wfl, float* __restrict__ out) {
    const int b    = blockIdx.x / EDGE_N;
    const int i    = blockIdx.x % EDGE_N;
    const int jt   = threadIdx.x >> 5;     // 0..9
    const int lane = threadIdx.x & 31;
    const int hf   = lane >> 4;
    const int lr   = lane & 15;

    __builtin_prefetch((const void*)(wfh + (size_t)lane * 16), 0, 1);

    const float* hbase = h + (size_t)b * EDGE_N * EDGE_E;
    const float* hi    = hbase + (size_t)i * EDGE_E;
    const int    jA    = jt * 16 + lr;                 // A-matrix row of this lane
    const float* hj    = hbase + (size_t)jA * EDGE_E;

    v8f acc[8];
    #pragma unroll
    for (int ot = 0; ot < 8; ++ot) {
        v8f z = {0.f, 0.f, 0.f, 0.f, 0.f, 0.f, 0.f, 0.f};
        acc[ot] = z;
    }

    // ---- edge-product GEMM: K = 128 in 4 bf16 K-steps, hi/lo split (x3) ----
    #pragma unroll
    for (int ks = 0; ks < 4; ++ks) {
        int c0 = ks * 32 + hf * 8, c1 = c0 + 16;
        v16bf ah, al;
        #pragma unroll
        for (int e = 0; e < 8; ++e) {
            float p0 = hj[c0 + e] * hi[c0 + e];
            float p1 = hj[c1 + e] * hi[c1 + e];
            __bf16 h0 = (__bf16)p0, h1 = (__bf16)p1;
            ah[e] = h0;     al[e]     = (__bf16)(p0 - (float)h0);
            ah[e + 8] = h1; al[e + 8] = (__bf16)(p1 - (float)h1);
        }
        #pragma unroll
        for (int ot = 0; ot < 8; ++ot) {
            size_t boff = ((size_t)(ks * 8 + ot) * 32 + lane) * 16;
            v16bf bh = *(const v16bf*)(wfh + boff);
            v16bf bl = *(const v16bf*)(wfl + boff);
            acc[ot] = __builtin_amdgcn_wmma_f32_16x16x32_bf16(false, ah, false, bh, (short)0, acc[ot], false, false);
            acc[ot] = __builtin_amdgcn_wmma_f32_16x16x32_bf16(false, ah, false, bl, (short)0, acc[ot], false, false);
            acc[ot] = __builtin_amdgcn_wmma_f32_16x16x32_bf16(false, al, false, bh, (short)0, acc[ot], false, false);
        }
    }

    // ---- spatial feats as one padded K-step: A2[j,s]=spat, B2=W_film[128+s]
    v16bf a2h, a2l;
    #pragma unroll
    for (int e = 0; e < 16; ++e) { a2h[e] = (__bf16)0.f; a2l[e] = (__bf16)0.f; }
    if (hf == 0) {                 // only lanes 0..15 carry K=0..7 (= s)
        const float* sprow = spat + (((size_t)b * EDGE_N + i) * EDGE_N + jA) * 8;
        #pragma unroll
        for (int e = 0; e < 8; ++e) {
            float v = sprow[e];
            __bf16 hv = (__bf16)v;
            a2h[e] = hv; a2l[e] = (__bf16)(v - (float)hv);
        }
    }
    #pragma unroll
    for (int ot = 0; ot < 8; ++ot) {
        int o = ot * 16 + lr;
        v16bf b2h, b2l;
        #pragma unroll
        for (int e = 0; e < 16; ++e) { b2h[e] = (__bf16)0.f; b2l[e] = (__bf16)0.f; }
        if (hf == 0) {
            #pragma unroll
            for (int e = 0; e < 8; ++e) {
                float v = Wf[(size_t)(EDGE_E + e) * EDGE_E + o];
                __bf16 hv = (__bf16)v;
                b2h[e] = hv; b2l[e] = (__bf16)(v - (float)hv);
            }
        }
        acc[ot] = __builtin_amdgcn_wmma_f32_16x16x32_bf16(false, a2h, false, b2h, (short)0, acc[ot], false, false);
        acc[ot] = __builtin_amdgcn_wmma_f32_16x16x32_bf16(false, a2h, false, b2l, (short)0, acc[ot], false, false);
        acc[ot] = __builtin_amdgcn_wmma_f32_16x16x32_bf16(false, a2l, false, b2h, (short)0, acc[ot], false, false);
        // + b_film (before LayerNorm, matching reference)
        float bo = bfm[o];
        #pragma unroll
        for (int r = 0; r < 8; ++r) acc[ot][r] += bo;
    }

    // ---- LayerNorm over the 128 outputs of each row ----
    // C layout: element (r,lane) -> row M=r+8*hf (cols in lanes of this half).
    float mu[8], rs[8];
    #pragma unroll
    for (int r = 0; r < 8; ++r) {
        float s1 = 0.f, s2 = 0.f;
        #pragma unroll
        for (int ot = 0; ot < 8; ++ot) { float v = acc[ot][r]; s1 += v; s2 += v * v; }
        #pragma unroll
        for (int m = 1; m <= 8; m <<= 1) {   // reduce within the 16-lane half
            s1 += __shfl_xor(s1, m, 32);
            s2 += __shfl_xor(s2, m, 32);
        }
        float mean = s1 * (1.f / 128.f);
        float var  = s2 * (1.f / 128.f) - mean * mean;
        mu[r] = mean;
        rs[r] = rsqrtf(var + 1e-5f);
    }

    // ---- FiLM + ReLU + store ----
    const float* gbb  = gb + (size_t)b * 256;
    float* obase = out + (((size_t)b * EDGE_N + i) * EDGE_N) * EDGE_E;
    #pragma unroll
    for (int ot = 0; ot < 8; ++ot) {
        int o = ot * 16 + lr;
        float g = gbb[o], be = gbb[EDGE_E + o];
        #pragma unroll
        for (int r = 0; r < 8; ++r) {
            int j = jt * 16 + r + 8 * hf;
            float v = (acc[ot][r] - mu[r]) * rs[r];
            v = g * v + be;
            obase[(size_t)j * EDGE_E + o] = v > 0.f ? v : 0.f;
        }
    }
}

// ---------------------------------------------------------------------------
extern "C" void kernel_launch(void* const* d_in, const int* in_sizes, int n_in,
                              void* d_out, int out_size, void* d_ws, size_t ws_size,
                              hipStream_t stream) {
    (void)in_sizes; (void)n_in; (void)out_size; (void)ws_size;
    const float* node = (const float*)d_in[0];   // [16,160,1024]
    const float* cond = (const float*)d_in[1];   // [16,1024]
    const float* spat = (const float*)d_in[2];   // [16,160,160,8]
    const float* Wp   = (const float*)d_in[3];   // [1024,128]
    const float* bp   = (const float*)d_in[4];   // [128]
    const float* Wc   = (const float*)d_in[5];   // [1024,256]
    const float* bc   = (const float*)d_in[6];   // [256]
    const float* Wf   = (const float*)d_in[7];   // [136,128]
    const float* bfm  = (const float*)d_in[8];   // [128]
    float* out = (float*)d_out;

    // workspace layout (all offsets 32B-aligned)
    char* ws = (char*)d_ws;
    float*  hbuf = (float*)(ws + 0);            // 2560*128 f32      = 1,310,720 B
    float*  gbuf = (float*)(ws + 1310720);      // 16*256 f32        =    16,384 B
    __bf16* wph  = (__bf16*)(ws + 1327104);     // 131072 bf16       =   262,144 B
    __bf16* wpl  = (__bf16*)(ws + 1589248);     // 131072 bf16
    __bf16* wfh  = (__bf16*)(ws + 1851392);     // 16384 bf16        =    32,768 B
    __bf16* wfl  = (__bf16*)(ws + 1884160);     // 16384 bf16

    k_pack_b<<<512, 256, 0, stream>>>(Wp, wph, wpl);   // K=1024: 131072 frags
    k_pack_b<<<64,  256, 0, stream>>>(Wf, wfh, wfl);   // K=128 (edge rows only)
    k_cond  <<<16,  256, 0, stream>>>(cond, Wc, bc, gbuf);
    k_proj  <<<160, 256, 0, stream>>>(node, bp, wph, wpl, hbuf);
    k_edge  <<<2560, 320, 0, stream>>>(hbuf, spat, Wf, bfm, gbuf, wfh, wfl, out);
}
